// LSTM_78005196030500
// MI455X (gfx1250) — matrix operand order repaired and checked by reading, not compiled, not run on hardware
//
#include <hip/hip_runtime.h>

// ---------------------------------------------------------------------------
// LSTM (VOCAB=50257, E=H=256, B=128, S=512) for gfx1250 / MI455X, wave32.
//
// Phase 0 (prep):   W_x, U_h, emb -> bf16 copies in ws; bias = b_x+b_u+b_g.
// Phase 1 (proj):   xp = bf16-WMMA( emb_bf16[tok], W_x^T ) + bias, stored f32
//                   in WMMA C-fragment order, software-pipelined loads,
//                   no in-loop conversions (emb pre-converted in prep).
// Phase 2 (scan):   8 workgroups (16 batch rows each), 512 sequential steps,
//                   h kept bf16 in LDS, c in registers, f32 accumulation,
//                   v_wmma_f32_16x16x32_bf16 for h @ U_h^T with double-
//                   buffered U_h fragments; xp prefetched one step ahead.
// ---------------------------------------------------------------------------

typedef __attribute__((ext_vector_type(16))) __bf16 v16bf;
typedef __attribute__((ext_vector_type(8)))  float  v8f;

#define VOCABN  50257
#define SEQL    512
#define BATCHN  128
#define EMBEDN  256
#define HIDN    256
#define GATESN  1024  // 4*HIDN

// workspace layout (bytes)
#define XP_ELEMS  ((size_t)SEQL * BATCHN * GATESN)          // 67,108,864 f32
#define WXB_OFF   (XP_ELEMS * 4)
#define UHB_OFF   (WXB_OFF + (size_t)GATESN * EMBEDN * 2)
#define BIAS_OFF  (UHB_OFF + (size_t)GATESN * HIDN * 2)
#define EMBB_OFF  (BIAS_OFF + (size_t)GATESN * 4)
// total: ~282 MiB

#define HPAD 296   // LDS row stride for h (ushort units): 592B = 37*16B, conflict-free

__device__ __forceinline__ unsigned short f2bf(float f) {
  unsigned u = __builtin_bit_cast(unsigned, f);
  u += 0x7fffu + ((u >> 16) & 1u);               // round-to-nearest-even
  return (unsigned short)(u >> 16);
}
__device__ __forceinline__ float fsig(float x) {
  return __builtin_amdgcn_rcpf(1.0f + __builtin_amdgcn_exp2f(x * -1.44269504f));
}
__device__ __forceinline__ float ftanh(float x) {
  return 2.0f * fsig(2.0f * x) - 1.0f;
}

union FragBF { v16bf v; unsigned short u[16]; unsigned d[8]; uint4 q[2]; };
union Frag32 { v8f v; float4 f4[2]; float f[8]; };

// --------------------------- Phase 0: prep ---------------------------------
// One pass: bf16 conversions for W_x, U_h and the whole embedding table,
// plus the fused bias. Pure streaming, fully parallel (~77 MB of traffic).
__global__ __launch_bounds__(256) void lstm_prep(
    const float* __restrict__ Wx, const float* __restrict__ Uh,
    const float* __restrict__ bx, const float* __restrict__ bu,
    const float* __restrict__ bg, const float* __restrict__ emb,
    unsigned short* __restrict__ Wbf, unsigned short* __restrict__ Ubf,
    float* __restrict__ bias, unsigned short* __restrict__ Ebf)
{
  const int i = blockIdx.x * 256 + threadIdx.x;
  if (i < GATESN * EMBEDN) {
    Wbf[i] = f2bf(Wx[i]);
    Ubf[i] = f2bf(Uh[i]);
  }
  if (i < GATESN) bias[i] = bx[i] + bu[i] + bg[i];
  if (i < VOCABN * EMBEDN) Ebf[i] = f2bf(emb[i]);
}

// ------------------- Phase 1: embed gather + input GEMM --------------------
// Rows ordered r = s*128 + b so each 16-row M-tile = (one s, one batch block).
// Grid: 2048 M-blocks (32 rows) x 4 N-blocks (256 cols) = 8192 WGs, 8 waves.
__global__ __launch_bounds__(256) void lstm_embed_proj(
    const int* __restrict__ tokens, const unsigned short* __restrict__ Ebf,
    const unsigned short* __restrict__ Wbf, const float* __restrict__ bias,
    float* __restrict__ xp)
{
  const int tid  = threadIdx.x;
  const int wv   = tid >> 5, lane = tid & 31;
  const int mblk = blockIdx.x >> 2, nblk = blockIdx.x & 3;
  const int wm   = wv & 1, wn = wv >> 1;
  const int Mt   = mblk * 2 + wm;          // 16-row tile index, 0..4095
  const int s    = Mt >> 3, bb = Mt & 7;
  const int m16  = lane & 15;
  const int hiA  = (lane >> 4) * 16;       // A-frag K sub-offset
  const int hiB  = (lane >> 4) * 16;       // B-frag K sub-offset
  const int b    = bb * 16 + m16;
  const int tok  = tokens[b * SEQL + s];
  const unsigned short* arow = Ebf + (size_t)tok * EMBEDN + hiA;

  v8f acc[4] = {};
  int ntg[4];
  const unsigned short* wbase[4];
#pragma unroll
  for (int tt = 0; tt < 4; ++tt) {
    ntg[tt]   = nblk * 16 + wn * 4 + tt;                          // 0..63
    wbase[tt] = Wbf + (size_t)(ntg[tt] * 16 + m16) * EMBEDN + hiB;
  }

  FragBF a[2], bf[2][4];

  // prologue: k-step 0 (A frag = two 16B chunks, no conversions)
  {
    a[0].q[0] = *(const uint4*)(arow);
    a[0].q[1] = *(const uint4*)(arow + 8);
#pragma unroll
    for (int tt = 0; tt < 4; ++tt) {
      const uint4* bp = (const uint4*)(wbase[tt]);
      bf[0][tt].q[0] = bp[0]; bf[0][tt].q[1] = bp[1];
    }
  }

#pragma unroll
  for (int kk = 0; kk < 8; ++kk) {
    const int cur = kk & 1, nxt = cur ^ 1;
    if (kk < 7) {   // issue next k-step's loads before consuming current
      a[nxt].q[0] = *(const uint4*)(arow + (kk + 1) * 32);
      a[nxt].q[1] = *(const uint4*)(arow + (kk + 1) * 32 + 8);
#pragma unroll
      for (int tt = 0; tt < 4; ++tt) {
        const uint4* bp = (const uint4*)(wbase[tt] + (kk + 1) * 32);
        bf[nxt][tt].q[0] = bp[0]; bf[nxt][tt].q[1] = bp[1];
      }
    }
#pragma unroll
    for (int tt = 0; tt < 4; ++tt)
      acc[tt] = __builtin_amdgcn_wmma_f32_16x16x32_bf16(
          false, a[cur].v, false, bf[cur][tt].v, (short)0, acc[tt], false, false);
  }

#pragma unroll
  for (int tt = 0; tt < 4; ++tt) {
    const float bn = bias[ntg[tt] * 16 + m16];   // C layout: N = lane&15
    Frag32 r;
#pragma unroll
    for (int e = 0; e < 8; ++e) r.f[e] = acc[tt][e] + bn;
    float4* dst = (float4*)(xp + (((size_t)(s * 8 + bb) * 64 + ntg[tt]) * 256) + lane * 8);
    dst[0] = r.f4[0];
    dst[1] = r.f4[1];
  }
}

// ------------------------- Phase 2: recurrent scan -------------------------
// blockIdx.x = batch block (16 rows). Wave w owns hidden units [32w,32w+32):
// its 8 N-tiles are nt = (t>>1)*16 + 2w + (t&1), so all four gates of a given
// (batch,hidden) land in the SAME lane -> pure in-register c/h update.
__global__ __launch_bounds__(256) void lstm_scan(
    const float* __restrict__ xp, const unsigned short* __restrict__ Ubf,
    const float* __restrict__ fcw, const float* __restrict__ fcb,
    float* __restrict__ out)
{
  __shared__ __align__(16) unsigned short h_lds[16 * HPAD];  // bf16 h
  __shared__ float h32[16 * 256];                            // final-step f32 h

  const int tid  = threadIdx.x;
  const int w    = tid >> 5, lane = tid & 31;
  const int bb   = blockIdx.x;
  const int m16  = lane & 15;
  const int hiA  = (lane >> 4) * 16;
  const int hiB  = (lane >> 4) * 16;
  const int mrow = (lane >> 4) * 8;   // C layout: M = e + 8*(lane>=16)

  // per-wave U_h fragment base pointers (constant across all 512 steps)
  const unsigned short* ubase[8];
#pragma unroll
  for (int t = 0; t < 8; ++t) {
    const int nt  = (t >> 1) * 16 + 2 * w + (t & 1);   // N-tile 0..63
    ubase[t] = Ubf + (size_t)(nt * 16 + m16) * HIDN + hiB;
  }

  for (int i = tid; i < 16 * HPAD; i += 256) h_lds[i] = 0;
  float c[16];
#pragma unroll
  for (int i = 0; i < 16; ++i) c[i] = 0.0f;
  __syncthreads();

#pragma unroll 1
  for (int s = 0; s < SEQL; ++s) {
    const float* xps = xp + ((size_t)(s * 8 + bb) * 64) * 256 + lane * 8;

    // init accumulators from xp (stored in fragment order -> coalesced b128s)
    Frag32 acc[8];
#pragma unroll
    for (int t = 0; t < 8; ++t) {
      const int nt = (t >> 1) * 16 + 2 * w + (t & 1);
      const float4* p = (const float4*)(xps + nt * 256);
      acc[t].f4[0] = p[0];
      acc[t].f4[1] = p[1];
    }

    // prefetch next step's xp fragments (global_prefetch, no counter cost)
    if (s < SEQL - 1) {
#pragma unroll
      for (int t = 0; t < 8; ++t) {
        const int nt = (t >> 1) * 16 + 2 * w + (t & 1);
        __builtin_prefetch(xps + (size_t)8 * 64 * 256 + nt * 256, 0, 0);
      }
    }

    // g += h @ U_h^T  (bf16 WMMA, f32 accumulate), K = 256 -> 8 k-tiles,
    // U_h fragments double-buffered so loads overlap the WMMAs.
    FragBF bfr[2][8];
#pragma unroll
    for (int t = 0; t < 8; ++t) {
      const uint4* bp = (const uint4*)(ubase[t]);
      bfr[0][t].q[0] = bp[0]; bfr[0][t].q[1] = bp[1];
    }
#pragma unroll
    for (int kk = 0; kk < 8; ++kk) {
      const int cur = kk & 1, nxt = cur ^ 1;
      if (kk < 7) {
#pragma unroll
        for (int t = 0; t < 8; ++t) {
          const uint4* bp = (const uint4*)(ubase[t] + (kk + 1) * 32);
          bfr[nxt][t].q[0] = bp[0]; bfr[nxt][t].q[1] = bp[1];
        }
      }
      FragBF a;
      const uint4* ap = (const uint4*)&h_lds[m16 * HPAD + kk * 32 + hiA];
      a.q[0] = ap[0];
      a.q[1] = ap[1];   // +8 bf16 elements = +1 uint4
#pragma unroll
      for (int t = 0; t < 8; ++t)
        acc[t].v = __builtin_amdgcn_wmma_f32_16x16x32_bf16(
            false, a.v, false, bfr[cur][t].v, (short)0, acc[t].v, false, false);
    }
    __syncthreads();   // all waves done reading h of step s-1

    // gate nonlinearities + state update, entirely per-lane
#pragma unroll
    for (int hf = 0; hf < 2; ++hf) {
      const int nh = 32 * w + 16 * hf + m16;     // hidden index
#pragma unroll
      for (int e = 0; e < 8; ++e) {
        const float gf = fsig (acc[0 + hf].f[e]);   // forget
        const float gi = fsig (acc[2 + hf].f[e]);   // input
        const float gc = ftanh(acc[4 + hf].f[e]);   // candidate
        const float go = fsig (acc[6 + hf].f[e]);   // output
        const float cn = gf * c[hf * 8 + e] + gi * gc;
        c[hf * 8 + e]  = cn;
        const float hv = go * ftanh(cn);
        const int m    = mrow + e;
        h_lds[m * HPAD + nh] = f2bf(hv);
        if (s == SEQL - 1) h32[m * 256 + nh] = hv;
      }
    }
    __syncthreads();   // new h visible before next step's reads
  }

  // fused final FC: out[b,:] = h @ fc_w^T + fc_b   (16 rows x 2 outs per WG)
  if (tid < 32) {
    const int b = tid >> 1, o = tid & 1;
    float sum = fcb[o];
    for (int n = 0; n < 256; ++n) sum += h32[b * 256 + n] * fcw[o * 256 + n];
    out[(bb * 16 + b) * 2 + o] = sum;
  }
}

// ---------------------------------------------------------------------------
extern "C" void kernel_launch(void* const* d_in, const int* in_sizes, int n_in,
                              void* d_out, int out_size, void* d_ws, size_t ws_size,
                              hipStream_t stream) {
  (void)in_sizes; (void)n_in; (void)out_size; (void)ws_size;
  const int*   tokens = (const int*)  d_in[0];
  const float* emb    = (const float*)d_in[1];
  const float* Wx     = (const float*)d_in[2];
  const float* bx     = (const float*)d_in[3];
  const float* Uh     = (const float*)d_in[4];
  const float* bu     = (const float*)d_in[5];
  const float* bg     = (const float*)d_in[6];
  const float* fcw    = (const float*)d_in[7];
  const float* fcb    = (const float*)d_in[8];
  float* out = (float*)d_out;

  char* ws = (char*)d_ws;
  float*          xp   = (float*)ws;                         // 256 MiB f32
  unsigned short* Wbf  = (unsigned short*)(ws + WXB_OFF);    // 512 KiB bf16
  unsigned short* Ubf  = (unsigned short*)(ws + UHB_OFF);    // 512 KiB bf16
  float*          bias = (float*)(ws + BIAS_OFF);            // 4 KiB f32
  unsigned short* Ebf  = (unsigned short*)(ws + EMBB_OFF);   // 24.5 MiB bf16

  lstm_prep      <<<VOCABN, 256, 0, stream>>>(Wx, Uh, bx, bu, bg, emb,
                                              Wbf, Ubf, bias, Ebf);
  lstm_embed_proj<<< 8192, 256, 0, stream>>>(tokens, Ebf, Wbf, bias, xp);
  lstm_scan      <<<    8, 256, 0, stream>>>(xp, Ubf, fcw, fcb, out);
}
